// ChannelAttention_45165876085428
// MI455X (gfx1250) — matrix operand order repaired and checked by reading, not compiled, not run on hardware
//
#include <hip/hip_runtime.h>
#include <hip/hip_fp16.h>

// ---------------------------------------------------------------------------
// ChannelAttention for MI455X (gfx1250, wave32).
// HBM-bound (~0.5 GB traffic @ 23.3 TB/s). Strategy:
//   - f16 intermediates + v_wmma_f32_16x16x32_f16 (fp32 accumulate)
//   - fuse projection with attn@V:  out = (P*blockdiag(attn)) * V + b
//   - async global->LDS copies (ASYNCcnt) with double-buffered LDS tiles
// ---------------------------------------------------------------------------

typedef __attribute__((ext_vector_type(16))) _Float16 v16h;
typedef __attribute__((ext_vector_type(8)))  float    v8f;
typedef __attribute__((ext_vector_type(4)))  int      v4i_t;

#define BATCH 16
#define CCH   512
#define NH    8
#define HD    64
#define HDIM  56
#define HW    3136            // 56*56 ; 3136 = 98*32 = 49*64

// ---- async global->LDS (gfx1250) with portable fallback -------------------
#if defined(__has_builtin)
#  if __has_builtin(__builtin_amdgcn_global_load_async_to_lds_b128) && \
      __has_builtin(__builtin_amdgcn_s_wait_asynccnt)
#    define ASYNC_OK 1
#  endif
#endif
#ifndef ASYNC_OK
#  define ASYNC_OK 0
#endif

#if ASYNC_OK
#  define WAIT_ASYNC(n) __builtin_amdgcn_s_wait_asynccnt(n)
#else
#  define WAIT_ASYNC(n)
#endif

// Copy 16 bytes global -> LDS. Both addresses must be 16B-aligned.
__device__ __forceinline__ void cp16(const _Float16* g, _Float16* l) {
#if ASYNC_OK
    // Builtin expects v4i addrspace(1)* (global src) and v4i addrspace(3)*
    // (LDS dst). addrspace(3) pointer = wave-relative LDS offset = low 32
    // bits of the generic LDS address (ISA 10.2 aperture mapping).
    __builtin_amdgcn_global_load_async_to_lds_b128(
        (__attribute__((address_space(1))) v4i_t*)(unsigned long long)(size_t)g,
        (__attribute__((address_space(3))) v4i_t*)(unsigned int)(size_t)l,
        0, 0);
#else
    *(float4*)l = *(const float4*)g;
#endif
}

// ---- WMMA fragment loaders (contiguous k = 16*half + s for A and B, wave32) -------------

// A (16x32 f16, row-major src): lane&15 = row M; pair i at K = kmapA(i,half)
__device__ __forceinline__ int kmapA(int i, int half) {
    return 16 * half + 2 * i;
}

template <typename PT>
__device__ __forceinline__ v16h load_frag_A(const PT* base, int ld, int lane) {
    int r = lane & 15, half = lane >> 4;
    v16h a;
#pragma unroll
    for (int i = 0; i < 8; ++i) {
        int k0 = kmapA(i, half);
        a[2 * i]     = (_Float16)base[r * ld + k0];
        a[2 * i + 1] = (_Float16)base[r * ld + k0 + 1];
    }
    return a;
}

// B (32x16 f16) from row-major K x N source: lane&15 = column N,
// VGPR i holds K rows 16*half + 2i, 2i+1.
template <typename PT>
__device__ __forceinline__ v16h load_frag_B(const PT* base, int ld, int lane) {
    int n = lane & 15, half = lane >> 4;
    v16h b;
#pragma unroll
    for (int i = 0; i < 8; ++i) {
        int k0 = 16 * half + 2 * i;
        b[2 * i]     = (_Float16)base[k0 * ld + n];
        b[2 * i + 1] = (_Float16)base[(k0 + 1) * ld + n];
    }
    return b;
}

// B operand taken as the TRANSPOSE of a row-major N x K source (for Q*K^T):
// lane&15 = column (row of src), contiguous 16 halves along K.
template <typename PT>
__device__ __forceinline__ v16h load_frag_BT(const PT* base, int ld, int lane) {
    int c = lane & 15, half = lane >> 4;
    v16h b;
#pragma unroll
    for (int i = 0; i < 8; ++i) {
        int k0 = 16 * half + 2 * i;
        b[2 * i]     = (_Float16)base[c * ld + k0];
        b[2 * i + 1] = (_Float16)base[c * ld + k0 + 1];
    }
    return b;
}

#define WMMA_F16(a, b, c) \
    __builtin_amdgcn_wmma_f32_16x16x32_f16(false, (a), false, (b), (short)0, (c), false, false)

// ---------------------------------------------------------------------------
// Stage A: depthwise 3x3 conv (group=C, 3 out/ch) + folded BN -> f16 q,k,v
// ---------------------------------------------------------------------------
__global__ void __launch_bounds__(256) dwconv_bn_qkv(
    const float* __restrict__ x, const float* __restrict__ w,
    const float* __restrict__ gamma, const float* __restrict__ beta,
    const float* __restrict__ mean, const float* __restrict__ var,
    _Float16* __restrict__ qkv) {
    int bo = blockIdx.x;                 // b*1536 + o
    int b  = bo / (3 * CCH);
    int o  = bo % (3 * CCH);
    int ic = o / 3;                      // group = out_channel / 3
    const float* xin = x + ((size_t)b * CCH + ic) * HW;

    float wr[9];
#pragma unroll
    for (int i = 0; i < 9; ++i) wr[i] = w[o * 9 + i];
    float inv   = gamma[o] * rsqrtf(var[o] + 1e-5f);
    float shift = beta[o] - mean[o] * inv;

    int s = o / CCH, c = o % CCH;        // qkv[s][b][c][n]
    _Float16* dst = qkv + (((size_t)s * BATCH + b) * CCH + c) * HW;

    for (int p = threadIdx.x; p < HW; p += blockDim.x) {
        int hh = p / HDIM, ww = p % HDIM;
        float acc = 0.f;
#pragma unroll
        for (int dy = -1; dy <= 1; ++dy) {
            int y = hh + dy;
            if (y < 0 || y >= HDIM) continue;
#pragma unroll
            for (int dx = -1; dx <= 1; ++dx) {
                int xx = ww + dx;
                if (xx < 0 || xx >= HDIM) continue;
                acc += xin[y * HDIM + xx] * wr[(dy + 1) * 3 + (dx + 1)];
            }
        }
        dst[p] = (_Float16)(acc * inv + shift);
    }
}

__global__ void cvt_proj(const float* __restrict__ src, _Float16* __restrict__ dst) {
    int i  = blockIdx.x * blockDim.x + threadIdx.x;
    dst[i] = (_Float16)src[i];
}

// ---------------------------------------------------------------------------
// Stage B: per (b,h): S = Q*K^T * scale over N=3136; softmax rows -> attn f16
// 4 waves/block; double-buffered async LDS tiles; each thread issues exactly
// 4 in-order async copies per tile -> s_wait_asynccnt(4) == prev tile landed.
// ---------------------------------------------------------------------------
__global__ void __launch_bounds__(128) attn_qk_softmax(
    const _Float16* __restrict__ qg, const _Float16* __restrict__ kg,
    _Float16* __restrict__ attn) {
    __shared__ _Float16 Qs[2][64 * 40];   // 40 halves = 80B row (16B multiple)
    __shared__ _Float16 Ks[2][64 * 40];
    __shared__ float    Ss[64 * 64];

    int bh = blockIdx.x;                 // b*8 + h
    int b = bh >> 3, h = bh & 7;
    const _Float16* Q = qg + ((size_t)b * CCH + h * HD) * HW;
    const _Float16* K = kg + ((size_t)b * CCH + h * HD) * HW;

    int tid = threadIdx.x, wave = tid >> 5, lane = tid & 31;
    v8f zero = {0.f, 0.f, 0.f, 0.f, 0.f, 0.f, 0.f, 0.f};
    v8f acc[4] = {zero, zero, zero, zero};

    // 64x32-half tile = 256 16B chunks; 2 chunks/thread/matrix.
    auto prefetch = [&](int buf, int n0) {
#pragma unroll
        for (int it = 0; it < 2; ++it) {
            int idx = tid + it * 128;
            int row = idx >> 2, c8 = (idx & 3) * 8;
            cp16(&Q[(size_t)row * HW + n0 + c8], &Qs[buf][row * 40 + c8]);
            cp16(&K[(size_t)row * HW + n0 + c8], &Ks[buf][row * 40 + c8]);
        }
    };

    int buf = 0;
    prefetch(0, 0);
    for (int kb = 0; kb < HW / 32; ++kb) {
        if (kb < HW / 32 - 1) {
            prefetch(buf ^ 1, (kb + 1) * 32);
            WAIT_ASYNC(4);               // tile kb complete (in-order)
        } else {
            WAIT_ASYNC(0);
        }
        __syncthreads();
        v16h a = load_frag_A(&Qs[buf][wave * 16 * 40], 40, lane);
#pragma unroll
        for (int t = 0; t < 4; ++t) {
            v16h bf = load_frag_BT(&Ks[buf][t * 16 * 40], 40, lane);  // K^T
            acc[t]  = WMMA_F16(a, bf, acc[t]);
        }
        __syncthreads();
        buf ^= 1;
    }

    int r = lane & 15, half = lane >> 4;
#pragma unroll
    for (int t = 0; t < 4; ++t)
#pragma unroll
        for (int j = 0; j < 8; ++j) {
            int d = wave * 16 + j + 8 * half;
            int e = t * 16 + r;
            Ss[d * 64 + e] = acc[t][j] * 0.125f;   // scale = hd^-0.5
        }
    __syncthreads();

    if (tid < 64) {                                  // one row per thread
        float* row = Ss + tid * 64;
        float m = row[0];
        for (int e = 1; e < 64; ++e) m = fmaxf(m, row[e]);
        float s = 0.f;
        for (int e = 0; e < 64; ++e) { float vv = __expf(row[e] - m); row[e] = vv; s += vv; }
        float is = 1.f / s;
        _Float16* dst = attn + (size_t)bh * 4096 + tid * 64;
        for (int e = 0; e < 64; ++e) dst[e] = (_Float16)(row[e] * is);
    }
}

// ---------------------------------------------------------------------------
// Stage C: M[b][c][(h,e)] = sum_d P[c][(h,d)] * attn[b,h][d][e]   (1 wave/blk)
// ---------------------------------------------------------------------------
__global__ void __launch_bounds__(32) build_M(
    const _Float16* __restrict__ P, const _Float16* __restrict__ attn,
    _Float16* __restrict__ Mg) {
    int id = blockIdx.x;
    int ct = id & 31;                    // 32 c-tiles of 16
    int h  = (id >> 5) & 7;
    int b  = id >> 8;
    int c0 = ct * 16;
    int lane = threadIdx.x;

    const _Float16* A  = P + (size_t)c0 * CCH + h * HD;            // 16 x 64
    const _Float16* At = attn + (size_t)(b * NH + h) * 4096;       // 64 x 64

    v8f zero = {0.f, 0.f, 0.f, 0.f, 0.f, 0.f, 0.f, 0.f};
    v8f acc[4] = {zero, zero, zero, zero};
#pragma unroll
    for (int kk = 0; kk < 2; ++kk) {
        v16h a = load_frag_A(A + kk * 32, CCH, lane);
#pragma unroll
        for (int t = 0; t < 4; ++t) {
            v16h bf = load_frag_B(At + (size_t)kk * 32 * 64 + t * 16, 64, lane);
            acc[t]  = WMMA_F16(a, bf, acc[t]);
        }
    }
    int r = lane & 15, half = lane >> 4;
    _Float16* Mb = Mg + (size_t)b * CCH * CCH;
#pragma unroll
    for (int t = 0; t < 4; ++t)
#pragma unroll
        for (int j = 0; j < 8; ++j)
            Mb[(size_t)(c0 + j + 8 * half) * CCH + h * HD + t * 16 + r] =
                (_Float16)acc[t][j];
}

// ---------------------------------------------------------------------------
// Stage D: out[b] = M_b (512x512) @ V_b (512x3136) + bias   (fp32 out, NCHW)
// 64x64 tile/block, 4 waves, double-buffered async LDS K-pipeline.
// ---------------------------------------------------------------------------
__global__ void __launch_bounds__(128) gemm_out(
    const _Float16* __restrict__ Mg, const _Float16* __restrict__ Vg,
    const float* __restrict__ bias, float* __restrict__ out) {
    __shared__ _Float16 Ms[2][64 * 40];   // 80B rows
    __shared__ _Float16 Vs[2][32 * 72];   // 144B rows (16B multiple)

    int n0 = blockIdx.x * 64, c0 = blockIdx.y * 64, b = blockIdx.z;
    const _Float16* Mbase = Mg + (size_t)b * CCH * CCH;
    const _Float16* Vbase = Vg + (size_t)b * CCH * HW;
    int tid = threadIdx.x, wave = tid >> 5, lane = tid & 31;

    v8f zero = {0.f, 0.f, 0.f, 0.f, 0.f, 0.f, 0.f, 0.f};
    v8f acc[4] = {zero, zero, zero, zero};

    // M tile 64x32 = 256 chunks (2/thread); V tile 32x64 = 256 chunks (2/thread)
    auto prefetch = [&](int buf, int k0) {
#pragma unroll
        for (int it = 0; it < 2; ++it) {
            int idx = tid + it * 128;
            int mr = idx >> 2, mc = (idx & 3) * 8;
            cp16(&Mbase[(size_t)(c0 + mr) * CCH + k0 + mc], &Ms[buf][mr * 40 + mc]);
            int vr = idx >> 3, vc = (idx & 7) * 8;
            cp16(&Vbase[(size_t)(k0 + vr) * HW + n0 + vc], &Vs[buf][vr * 72 + vc]);
        }
    };

    int buf = 0;
    prefetch(0, 0);
    for (int kk = 0; kk < CCH / 32; ++kk) {
        if (kk < CCH / 32 - 1) {
            prefetch(buf ^ 1, (kk + 1) * 32);
            WAIT_ASYNC(4);
        } else {
            WAIT_ASYNC(0);
        }
        __syncthreads();
        v16h a = load_frag_A(&Ms[buf][wave * 16 * 40], 40, lane);
#pragma unroll
        for (int t = 0; t < 4; ++t) {
            v16h bf = load_frag_B(&Vs[buf][t * 16], 72, lane);
            acc[t]  = WMMA_F16(a, bf, acc[t]);
        }
        __syncthreads();
        buf ^= 1;
    }

    int r = lane & 15, half = lane >> 4;
#pragma unroll
    for (int t = 0; t < 4; ++t)
#pragma unroll
        for (int j = 0; j < 8; ++j) {
            int c = c0 + wave * 16 + j + 8 * half;
            int n = n0 + t * 16 + r;
            out[((size_t)b * CCH + c) * HW + n] = acc[t][j] + bias[c];
        }
}

// ---------------------------------------------------------------------------
extern "C" void kernel_launch(void* const* d_in, const int* in_sizes, int n_in,
                              void* d_out, int out_size, void* d_ws, size_t ws_size,
                              hipStream_t stream) {
    (void)in_sizes; (void)n_in; (void)out_size; (void)ws_size;
    const float* x        = (const float*)d_in[0];
    const float* conv_w   = (const float*)d_in[1];
    const float* bn_gamma = (const float*)d_in[2];
    const float* bn_beta  = (const float*)d_in[3];
    const float* bn_mean  = (const float*)d_in[4];
    const float* bn_var   = (const float*)d_in[5];
    const float* proj_w   = (const float*)d_in[6];
    const float* proj_b   = (const float*)d_in[7];
    float* out = (float*)d_out;

    // Workspace layout (f16): q|k|v (3*16*512*3136), attn (16*8*64*64),
    // proj16 (512*512), M (16*512*512)  -> ~164 MB total.
    const size_t QKV1 = (size_t)BATCH * CCH * HW;
    _Float16* qkv    = (_Float16*)d_ws;
    _Float16* q      = qkv;
    _Float16* k      = qkv + QKV1;
    _Float16* v      = qkv + 2 * QKV1;
    _Float16* attn   = qkv + 3 * QKV1;
    _Float16* proj16 = attn + (size_t)BATCH * NH * 64 * 64;
    _Float16* M      = proj16 + (size_t)CCH * CCH;

    dwconv_bn_qkv<<<BATCH * 3 * CCH, 256, 0, stream>>>(
        x, conv_w, bn_gamma, bn_beta, bn_mean, bn_var, qkv);
    cvt_proj<<<(CCH * CCH) / 256, 256, 0, stream>>>(proj_w, proj16);
    attn_qk_softmax<<<BATCH * NH, 128, 0, stream>>>(q, k, attn);
    build_M<<<BATCH * NH * (CCH / 16), 32, 0, stream>>>(proj16, attn, M);
    gemm_out<<<dim3(HW / 64, CCH / 64, BATCH), 128, 0, stream>>>(M, v, proj_b, out);
}
